// AIF_62448824484127
// MI455X (gfx1250) — matrix operand order repaired and checked by reading, not compiled, not run on hardware
//
#include <hip/hip_runtime.h>
#include <hip/hip_bf16.h>
#include <math.h>

// ---- problem constants ----
#define BB 16
#define TT 2048
#define VV 256
#define DD 1024

typedef __attribute__((ext_vector_type(16))) __bf16 v16bf;
typedef __attribute__((ext_vector_type(8)))  __bf16 v8bf;
typedef __attribute__((ext_vector_type(8)))  float  v8f;
typedef __attribute__((ext_vector_type(4)))  float  v4f;

#define WMMA_BF16(a, b, c) \
    __builtin_amdgcn_wmma_f32_16x16x32_bf16(false, (a), false, (b), (short)0, (c), false, false)

// ---------------------------------------------------------------------------
// K1: alphas = sigmoid(xs . W_out + b_out) * (t < xs_len)
// one wave per (b,t); 8 waves / block
// ---------------------------------------------------------------------------
__global__ void aif_alpha_kernel(const float* __restrict__ xs,
                                 const int*   __restrict__ xs_lens,
                                 const float* __restrict__ W_out,
                                 const float* __restrict__ b_out,
                                 float* __restrict__ alphas) {
    int wid  = blockIdx.x * (blockDim.x >> 5) + (threadIdx.x >> 5);
    int lane = threadIdx.x & 31;
    int b = wid >> 11;          // T = 2048
    int t = wid & (TT - 1);

    const v4f* xrow = (const v4f*)(xs + (size_t)wid * DD);
    const v4f* w4   = (const v4f*)W_out;
    float dot = 0.f;
#pragma unroll
    for (int j = 0; j < 8; ++j) {
        v4f xv = xrow[j * 32 + lane];
        v4f wv = w4[j * 32 + lane];
        dot += xv.x * wv.x + xv.y * wv.y + xv.z * wv.z + xv.w * wv.w;
    }
#pragma unroll
    for (int off = 16; off > 0; off >>= 1)
        dot += __shfl_xor(dot, off, 32);

    if (lane == 0) {
        float a = 1.0f / (1.0f + expf(-(dot + b_out[0])));
        if (t >= xs_lens[b]) a = 0.0f;
        alphas[wid] = a;
    }
}

// ---------------------------------------------------------------------------
// K2: per-batch inclusive cumsum of alphas -> csum; total -> token_num
// ---------------------------------------------------------------------------
__global__ void aif_scan_kernel(const float* __restrict__ alphas,
                                float* __restrict__ csum,
                                float* __restrict__ token_num) {
    __shared__ float s[256];
    int tid = threadIdx.x;
    size_t base = (size_t)blockIdx.x * TT;

    float l[8];
    float run = 0.f;
#pragma unroll
    for (int i = 0; i < 8; ++i) {
        l[i] = alphas[base + tid * 8 + i];
        run += l[i];
        l[i] = run;                       // inclusive within chunk
    }
    s[tid] = run;
    __syncthreads();
    for (int off = 1; off < 256; off <<= 1) {
        float v = (tid >= off) ? s[tid - off] : 0.f;
        __syncthreads();
        s[tid] += v;
        __syncthreads();
    }
    float excl = s[tid] - run;
#pragma unroll
    for (int i = 0; i < 8; ++i)
        csum[base + tid * 8 + i] = excl + l[i];
    if (tid == 255) token_num[blockIdx.x] = s[255];
}

// ---------------------------------------------------------------------------
// K3: repack W_proj into B-fragment order (bf16), contiguous k = 16*(lane/16) + e as in the A loads
// ---------------------------------------------------------------------------
__global__ void aif_repack_w_kernel(const float* __restrict__ W,
                                    __bf16* __restrict__ wfrag) {
    int ntile = blockIdx.x >> 5;
    int ks    = blockIdx.x & 31;
    int lane  = threadIdx.x;
    int n  = ntile * 16 + (lane & 15);
    int kb = (lane >= 16) ? 16 : 0;
    v16bf f;
#pragma unroll
    for (int e = 0; e < 16; ++e) {
        int k = ks * 32 + kb + e;
        f[e] = (__bf16)W[(size_t)k * DD + n];
    }
    ((v16bf*)wfrag)[(size_t)blockIdx.x * 32 + lane] = f;
}

// A-fragment from a row-major f32 row: one contiguous run of 16 floats
__device__ inline v16bf load_a_frag_f32(const float* row, int kbase) {
    v4f r0 = *(const v4f*)(row + kbase);
    v4f r1 = *(const v4f*)(row + kbase + 4);
    v4f r2 = *(const v4f*)(row + kbase + 8);
    v4f r3 = *(const v4f*)(row + kbase + 12);
    v16bf a;
    a[0]  = (__bf16)r0.x; a[1]  = (__bf16)r0.y; a[2]  = (__bf16)r0.z; a[3]  = (__bf16)r0.w;
    a[4]  = (__bf16)r1.x; a[5]  = (__bf16)r1.y; a[6]  = (__bf16)r1.z; a[7]  = (__bf16)r1.w;
    a[8]  = (__bf16)r2.x; a[9]  = (__bf16)r2.y; a[10] = (__bf16)r2.z; a[11] = (__bf16)r2.w;
    a[12] = (__bf16)r3.x; a[13] = (__bf16)r3.y; a[14] = (__bf16)r3.z; a[15] = (__bf16)r3.w;
    return a;
}

// A-fragment from a row-major bf16 row
__device__ inline v16bf load_a_frag_bf16(const __bf16* row, int kbase) {
    v8bf a0 = *(const v8bf*)(row + kbase);
    v8bf a1 = *(const v8bf*)(row + kbase + 8);
    v16bf a;
#pragma unroll
    for (int e = 0; e < 8; ++e) { a[e] = a0[e]; a[8 + e] = a1[e]; }
    return a;
}

// ---------------------------------------------------------------------------
// K4: xs_proj = xs @ W_proj + b_proj  -> bf16 in (t,d) and (d,t) layouts
// M = 32768, N = 1024, K = 1024.  2(M) x 4(N) 16x16 tiles per wave.
// ---------------------------------------------------------------------------
__global__ void aif_gemm1_kernel(const float*  __restrict__ xs,
                                 const __bf16* __restrict__ wfrag,
                                 const float*  __restrict__ b_proj,
                                 __bf16* __restrict__ xsp,     // (B*T, D)
                                 __bf16* __restrict__ xspT) {  // (B, D, T)
    int wid  = blockIdx.x * 8 + (threadIdx.x >> 5);
    int lane = threadIdx.x & 31;
    int mg = wid >> 4;            // 1024 groups of 32 rows
    int ng = wid & 15;            // 16 groups of 64 cols

    int kb = (lane >= 16) ? 16 : 0;
    const float* arow0 = xs + (size_t)(mg * 32 + (lane & 15)) * DD;
    const float* arow1 = arow0 + (size_t)16 * DD;
    const v16bf* wf = (const v16bf*)wfrag;

    v8f acc[2][4];
#pragma unroll
    for (int i = 0; i < 2; ++i)
#pragma unroll
        for (int j = 0; j < 4; ++j)
            acc[i][j] = (v8f){0.f, 0.f, 0.f, 0.f, 0.f, 0.f, 0.f, 0.f};

    for (int ks = 0; ks < DD / 32; ++ks) {
        v16bf a0 = load_a_frag_f32(arow0, ks * 32 + kb);
        v16bf a1 = load_a_frag_f32(arow1, ks * 32 + kb);
        if (ks < 28) {  // keep the A streams ahead in cache
            __builtin_prefetch(arow0 + ks * 32 + 128, 0, 3);
            __builtin_prefetch(arow1 + ks * 32 + 128, 0, 3);
        }
#pragma unroll
        for (int j = 0; j < 4; ++j) {
            v16bf b = wf[((size_t)(ng * 4 + j) * 32 + ks) * 32 + lane];
            acc[0][j] = WMMA_BF16(a0, b, acc[0][j]);
            acc[1][j] = WMMA_BF16(a1, b, acc[1][j]);
        }
    }

    int moff = (lane >= 16) ? 8 : 0;
#pragma unroll
    for (int j = 0; j < 4; ++j) {
        int n = (ng * 4 + j) * 16 + (lane & 15);
        float bias = b_proj[n];
#pragma unroll
        for (int i = 0; i < 2; ++i) {
            int mbase = mg * 32 + i * 16;
#pragma unroll
            for (int r = 0; r < 8; ++r) {
                int m = mbase + r + moff;
                __bf16 bv = (__bf16)(acc[i][j][r] + bias);
                xsp[(size_t)m * DD + n] = bv;
                int b_ = m >> 11;              // T = 2048
                int t_ = m & (TT - 1);
                xspT[((size_t)b_ * DD + n) * TT + t_] = bv;
            }
        }
    }
}

// ---------------------------------------------------------------------------
// K5: scores = (ys @ xs_proj^T) * scale, AIF mask fused -> f32
// per batch: M = V = 256, N = T = 2048, K = D = 1024.  2x4 tiles per wave.
// ---------------------------------------------------------------------------
__global__ void aif_gemm2_kernel(const float*  __restrict__ ys,
                                 const __bf16* __restrict__ xsp,
                                 const float*  __restrict__ csum,
                                 const int*    __restrict__ xs_lens,
                                 const int*    __restrict__ ys_lens,
                                 float* __restrict__ scores) {
    int wid  = blockIdx.x * 8 + (threadIdx.x >> 5);
    int lane = threadIdx.x & 31;
    int b   = wid >> 8;            // 8 vgroups * 32 tgroups per batch
    int rem = wid & 255;
    int vg  = rem >> 5;            // 32 rows of V
    int tg  = rem & 31;            // 64 cols of T

    int kb = (lane >= 16) ? 16 : 0;
    int tb = (lane >= 16) ? 16 : 0;
    const float* arow0 = ys + ((size_t)b * VV + vg * 32 + (lane & 15)) * DD;
    const float* arow1 = arow0 + (size_t)16 * DD;
    const __bf16* brow[4];
#pragma unroll
    for (int j = 0; j < 4; ++j)
        brow[j] = xsp + ((size_t)b * TT + (tg * 4 + j) * 16 + (lane & 15)) * DD;

    v8f acc[2][4];
#pragma unroll
    for (int i = 0; i < 2; ++i)
#pragma unroll
        for (int j = 0; j < 4; ++j)
            acc[i][j] = (v8f){0.f, 0.f, 0.f, 0.f, 0.f, 0.f, 0.f, 0.f};

    for (int ks = 0; ks < DD / 32; ++ks) {
        v16bf a0 = load_a_frag_f32(arow0, ks * 32 + kb);
        v16bf a1 = load_a_frag_f32(arow1, ks * 32 + kb);
#pragma unroll
        for (int j = 0; j < 4; ++j) {
            v16bf bf = *(const v16bf*)(brow[j] + ks * 32 + tb);
            acc[0][j] = WMMA_BF16(a0, bf, acc[0][j]);
            acc[1][j] = WMMA_BF16(a1, bf, acc[1][j]);
        }
    }

    int xl = xs_lens[b];
    int yl = ys_lens[b];
    int moff = (lane >= 16) ? 8 : 0;
    const float scale = 0.03125f;      // 1/sqrt(1024)
#pragma unroll
    for (int j = 0; j < 4; ++j) {
        int t = (tg * 4 + j) * 16 + (lane & 15);
        float cs = csum[(size_t)b * TT + t];
        bool tvalid = (t < xl);
#pragma unroll
        for (int i = 0; i < 2; ++i) {
#pragma unroll
            for (int r = 0; r < 8; ++r) {
                int v = vg * 32 + i * 16 + r + moff;
                bool m = tvalid && (v < yl) && (cs < (float)(v + 1));
                float sv = m ? acc[i][j][r] * scale : -1e9f;
                scores[((size_t)b * VV + v) * TT + t] = sv;
            }
        }
    }
}

// ---------------------------------------------------------------------------
// K6: softmax over T per (b,v) row -> attn (bf16)
// ---------------------------------------------------------------------------
__global__ void aif_softmax_kernel(const float* __restrict__ scores,
                                   __bf16* __restrict__ attn) {
    __shared__ float red[256];
    int tid = threadIdx.x;
    size_t base = (size_t)blockIdx.x * TT;

    float s[8];
    float mx = -3.4e38f;
#pragma unroll
    for (int i = 0; i < 8; ++i) {
        s[i] = scores[base + tid + i * 256];
        mx = fmaxf(mx, s[i]);
    }
    red[tid] = mx; __syncthreads();
    for (int off = 128; off > 0; off >>= 1) {
        if (tid < off) red[tid] = fmaxf(red[tid], red[tid + off]);
        __syncthreads();
    }
    mx = red[0]; __syncthreads();

    float se = 0.f;
#pragma unroll
    for (int i = 0; i < 8; ++i) { s[i] = expf(s[i] - mx); se += s[i]; }
    red[tid] = se; __syncthreads();
    for (int off = 128; off > 0; off >>= 1) {
        if (tid < off) red[tid] += red[tid + off];
        __syncthreads();
    }
    float inv = 1.0f / red[0];
#pragma unroll
    for (int i = 0; i < 8; ++i)
        attn[base + tid + i * 256] = (__bf16)(s[i] * inv);
}

// ---------------------------------------------------------------------------
// K7: acoustic = attn @ xs_proj -> f32 output
// per batch: M = V = 256, N = D = 1024, K = T = 2048.  2x4 tiles per wave.
// ---------------------------------------------------------------------------
__global__ void aif_gemm3_kernel(const __bf16* __restrict__ attn,
                                 const __bf16* __restrict__ xspT, // (B,D,T)
                                 float* __restrict__ out) {       // (B,V,D)
    int wid  = blockIdx.x * 8 + (threadIdx.x >> 5);
    int lane = threadIdx.x & 31;
    int b   = wid >> 7;            // 8 vgroups * 16 dgroups per batch
    int rem = wid & 127;
    int vg  = rem >> 4;            // 32 rows of V
    int dg  = rem & 15;            // 64 cols of D

    int kb = (lane >= 16) ? 16 : 0;
    int tb = (lane >= 16) ? 16 : 0;
    const __bf16* arow0 = attn + ((size_t)b * VV + vg * 32 + (lane & 15)) * TT;
    const __bf16* arow1 = arow0 + (size_t)16 * TT;
    const __bf16* brow[4];
#pragma unroll
    for (int j = 0; j < 4; ++j)
        brow[j] = xspT + ((size_t)b * DD + (dg * 4 + j) * 16 + (lane & 15)) * TT;

    v8f acc[2][4];
#pragma unroll
    for (int i = 0; i < 2; ++i)
#pragma unroll
        for (int j = 0; j < 4; ++j)
            acc[i][j] = (v8f){0.f, 0.f, 0.f, 0.f, 0.f, 0.f, 0.f, 0.f};

    for (int ks = 0; ks < TT / 32; ++ks) {
        v16bf a0 = load_a_frag_bf16(arow0, ks * 32 + kb);
        v16bf a1 = load_a_frag_bf16(arow1, ks * 32 + kb);
        if (ks < 56) {
            __builtin_prefetch(arow0 + ks * 32 + 256, 0, 3);
            __builtin_prefetch(arow1 + ks * 32 + 256, 0, 3);
        }
#pragma unroll
        for (int j = 0; j < 4; ++j) {
            v16bf bf = *(const v16bf*)(brow[j] + ks * 32 + tb);
            acc[0][j] = WMMA_BF16(a0, bf, acc[0][j]);
            acc[1][j] = WMMA_BF16(a1, bf, acc[1][j]);
        }
    }

    int moff = (lane >= 16) ? 8 : 0;
#pragma unroll
    for (int j = 0; j < 4; ++j) {
        int d = (dg * 4 + j) * 16 + (lane & 15);
#pragma unroll
        for (int i = 0; i < 2; ++i) {
#pragma unroll
            for (int r = 0; r < 8; ++r) {
                int v = vg * 32 + i * 16 + r + moff;
                out[((size_t)b * VV + v) * DD + d] = acc[i][j][r];
            }
        }
    }
}

// ---------------------------------------------------------------------------
extern "C" void kernel_launch(void* const* d_in, const int* in_sizes, int n_in,
                              void* d_out, int out_size, void* d_ws, size_t ws_size,
                              hipStream_t stream) {
    (void)in_sizes; (void)n_in; (void)out_size; (void)ws_size;

    const float* xs      = (const float*)d_in[0];
    const int*   xs_lens = (const int*)  d_in[1];
    const float* ys      = (const float*)d_in[2];
    const int*   ys_lens = (const int*)  d_in[3];
    const float* W_out   = (const float*)d_in[4];
    const float* b_out   = (const float*)d_in[5];
    const float* W_proj  = (const float*)d_in[6];
    const float* b_proj  = (const float*)d_in[7];

    // d_out: acoustic (B*V*D) | token_num (B) | alphas (B*T) | csum (B*T)
    float* out_ac    = (float*)d_out;
    float* out_tok   = out_ac  + (size_t)BB * VV * DD;
    float* out_alpha = out_tok + BB;
    float* out_csum  = out_alpha + (size_t)BB * TT;

    // workspace layout (bytes)
    char* ws = (char*)d_ws;
    __bf16* wfrag = (__bf16*)(ws);                                   //   2 MB
    __bf16* xsp   = (__bf16*)(ws + (size_t)2097152);                 //  64 MB
    __bf16* xspT  = (__bf16*)(ws + (size_t)2097152 + 67108864);      //  64 MB
    float*  scr   = (float*) (ws + (size_t)2097152 + 2*67108864);    //  32 MB
    __bf16* attn  = (__bf16*)(ws + (size_t)2097152 + 2*67108864 + 33554432); // 16 MB

    // K1: alphas  (B*T waves, 8 waves / block)
    aif_alpha_kernel<<<(BB * TT) / 8, 256, 0, stream>>>(xs, xs_lens, W_out, b_out, out_alpha);
    // K2: cumsum + token_num
    aif_scan_kernel<<<BB, 256, 0, stream>>>(out_alpha, out_csum, out_tok);
    // K3: repack W_proj into B-fragment order
    aif_repack_w_kernel<<<(DD / 16) * (DD / 32), 32, 0, stream>>>(W_proj, wfrag);
    // K4: xs_proj GEMM  (1024 mgroups * 16 ngroups waves)
    aif_gemm1_kernel<<<(1024 * 16) / 8, 256, 0, stream>>>(xs, wfrag, b_proj, xsp, xspT);
    // K5: scores GEMM + AIF mask  (16 * 8 * 32 waves)
    aif_gemm2_kernel<<<(BB * 8 * 32) / 8, 256, 0, stream>>>(ys, xsp, out_csum,
                                                            xs_lens, ys_lens, scr);
    // K6: softmax rows
    aif_softmax_kernel<<<BB * VV, 256, 0, stream>>>(scr, attn);
    // K7: output GEMM  (16 * 8 * 16 waves)
    aif_gemm3_kernel<<<(BB * 8 * 16) / 8, 256, 0, stream>>>(attn, xspT, out_ac);
}
